// WeightAdaptiveGallinear_12163347382922
// MI455X (gfx1250) — compile-verified, matrix operand order checked by reading, not run on hardware
//
#include <hip/hip_runtime.h>
#include <hip/hip_bf16.h>
#include <math.h>

// Problem constants from the reference
#define BATCH   65536
#define XCOLS   130      // IN + 1 + LAT
#define LAT     128
#define HID     2048
#define OUTW    512
#define NFOLD   1024     // (IN+1)*OUT after folding v

typedef __attribute__((ext_vector_type(16))) __bf16 v16bf;
typedef __attribute__((ext_vector_type(8)))  __bf16 v8bf;
typedef __attribute__((ext_vector_type(8)))  float  v8f;

static __device__ __forceinline__ v16bf cat8(v8bf lo, v8bf hi) {
    return __builtin_shufflevector(lo, hi, 0,1,2,3,4,5,6,7,8,9,10,11,12,13,14,15);
}

static __device__ __forceinline__ v8f wmma_bf16(v16bf a, v16bf b, v8f c) {
    // emits v_wmma_f32_16x16x32_bf16
    return __builtin_amdgcn_wmma_f32_16x16x32_bf16(false, a, false, b, (short)0, c, false, false);
}

// Branch-free tanh: hardware v_tanh_f32 if available, else exp2+rcp identity.
static __device__ __forceinline__ float fast_tanh(float x) {
#if __has_builtin(__builtin_amdgcn_tanhf)
    return __builtin_amdgcn_tanhf(x);
#else
    float ax = __builtin_fabsf(x);
    float e  = __builtin_amdgcn_exp2f(-2.88539008178f * ax);   // exp(-2|x|)
    float t  = (1.0f - e) * __builtin_amdgcn_rcpf(1.0f + e);   // tanh(|x|)
    return __builtin_copysignf(t, x);
#endif
}

// ---------------- Prep: compute scalar v0,v1 once ----------------
__global__ void compute_v_kernel(const float* __restrict__ x, const float* __restrict__ dil,
                                 const float* __restrict__ sp, float* __restrict__ vout) {
    if (threadIdx.x != 0 || blockIdx.x != 0) return;
    float s  = x[(size_t)(BATCH - 1) * XCOLS + 1];   // x[-1, IN]
    float t0 = s * dil[0] + sp[0];
    float t1 = s * dil[1] + sp[1];
    float t2 = s * dil[2] + sp[2];
    float t3 = s * dil[3] + sp[3];
    vout[0] = cosf(t0) + cosf(t2);
    vout[1] = sinf(t1) + sinf(t3);
}

// ---------------- Prep kernels: pack weights into WMMA B-fragment order ---------------
// Packed layout for a KxN matrix: tiles of (32 x 16). Element index:
//   idx = (tileLinear*512) + lane*16 + e, lane = 0..31, e = 0..15
//   g = lane>>4, n_in_tile = lane&15
//   k_in_tile = (e<8) ? 8*g + e : 16 + 8*g + (e-8)
// so each lane's 16 bf16 values (32 bytes) are contiguous -> 2x global_load_b128.

__global__ void pack_w1_kernel(const float* __restrict__ w1, __bf16* __restrict__ outp) {
    int idx = blockIdx.x * 256 + threadIdx.x;          // 128*2048 = 262144 total
    if (idx >= LAT * HID) return;
    int e    = idx & 15;
    int L    = (idx >> 4) & 31;
    int tile = idx >> 9;
    int kt   = tile & 3;                               // 128/32 = 4 K-tiles
    int nt   = tile >> 2;                              // 2048/16 = 128 N-tiles
    int g    = L >> 4;
    int kl   = (e < 8) ? (8 * g + e) : (16 + 8 * g + (e - 8));
    int k    = kt * 32 + kl;
    int n    = nt * 16 + (L & 15);
    outp[idx] = (__bf16)w1[(size_t)k * HID + n];
}

__global__ void pack_w2_kernel(const float* __restrict__ w2, const float* __restrict__ vbuf,
                               __bf16* __restrict__ outp) {
    int idx = blockIdx.x * 256 + threadIdx.x;          // 2048*1024 = 2097152 total
    if (idx >= HID * NFOLD) return;
    float v0 = vbuf[0], v1 = vbuf[1];
    int e    = idx & 15;
    int L    = (idx >> 4) & 31;
    int tile = idx >> 9;
    int kt   = tile & 63;                              // 2048/32 = 64 K-tiles
    int nt   = tile >> 6;                              // 1024/16 = 64 N-tiles
    int g    = L >> 4;
    int kl   = (e < 8) ? (8 * g + e) : (16 + 8 * g + (e - 8));
    int k    = kt * 32 + kl;
    int j    = nt * 16 + (L & 15);
    const float* rowk = w2 + (size_t)k * (2 * NFOLD);
    outp[idx] = (__bf16)(rowk[2 * j] * v0 + rowk[2 * j + 1] * v1);
}

__global__ void fold_b2_kernel(const float* __restrict__ b2in, const float* __restrict__ vbuf,
                               float* __restrict__ b2out) {
    int j = blockIdx.x * 256 + threadIdx.x;
    if (j >= NFOLD) return;
    b2out[j] = b2in[2 * j] * vbuf[0] + b2in[2 * j + 1] * vbuf[1];
}

// ---------------- Fused main kernel ----------------
// Block = 256 threads (8 waves), handles 64 batch rows.
// LDS: z (64x136 bf16, padded), inp (64 f32), h (64x2056 bf16, padded).
#define Z_STRIDE 136
#define H_STRIDE 2056
#define Z_BYTES  (64 * Z_STRIDE * 2)            // 17408
#define INP_OFF  Z_BYTES                        // 17408
#define H_OFF    (INP_OFF + 256)                // 17664
#define SMEM_BYTES (H_OFF + 64 * H_STRIDE * 2)  // 280832

__global__ __launch_bounds__(256)
void fused_gal_kernel(const float* __restrict__ x,
                      const __bf16* __restrict__ w1p, const float* __restrict__ b1,
                      const __bf16* __restrict__ w2p, const float* __restrict__ b2,
                      float* __restrict__ out) {
    extern __shared__ __align__(16) char smem[];
    __bf16* zS  = (__bf16*)smem;
    float*  inpS = (float*)(smem + INP_OFF);
    __bf16* hS  = (__bf16*)(smem + H_OFF);

    const int tid    = threadIdx.x;
    const int wgBase = blockIdx.x * 64;

    // Stage z (bf16) and inp (f32) into LDS
    for (int i = tid; i < 64 * LAT; i += 256) {
        int r = i >> 7, c = i & 127;
        zS[r * Z_STRIDE + c] = (__bf16)x[(size_t)(wgBase + r) * XCOLS + 2 + c];
    }
    if (tid < 64) inpS[tid] = x[(size_t)(wgBase + tid) * XCOLS];
    __syncthreads();

    const int wave = tid >> 5, lane = tid & 31;
    const int mt   = wave >> 1;        // M-tile 0..3 (16 rows each)
    const int half = wave & 1;         // N split
    const int g    = lane >> 4, ln = lane & 15;

    // ---- GEMM1: h = tanh(z @ W1 + b1), K = 128 (4 WMMA steps) ----
    v16bf a1[4];
    {
        const __bf16* rowp = zS + (mt * 16 + ln) * Z_STRIDE;
#pragma unroll
        for (int kt = 0; kt < 4; ++kt) {
            const __bf16* p = rowp + kt * 32 + 8 * g;
            a1[kt] = cat8(*(const v8bf*)p, *(const v8bf*)(p + 16));
        }
    }

    for (int nt2 = 0; nt2 < 64; ++nt2) {
        int nt = half * 64 + nt2;
        v8f acc = {};
        const __bf16* bp = w1p + (size_t)nt * 4 * 512 + lane * 16;
#pragma unroll
        for (int kt = 0; kt < 4; ++kt) {
            v16bf b = *(const v16bf*)(bp + kt * 512);
            acc = wmma_bf16(a1[kt], b, acc);
        }
        int   col  = nt * 16 + ln;
        float bias = b1[col];
        int   rbase = mt * 16 + 8 * g;
#pragma unroll
        for (int r = 0; r < 8; ++r) {
            float v = fast_tanh(acc[r] + bias);
            hS[(rbase + r) * H_STRIDE + col] = (__bf16)v;
        }
    }
    __syncthreads();

    // ---- GEMM2: w = h @ W2f + b2f ; combine out = inp*w[:,o] + w[:,512+o] ----
    // wave covers 16 paired N-tiles: (ntLo, ntLo+32); process 4 pairs per K-sweep.
    for (int pg = 0; pg < 4; ++pg) {
        v8f accLo[4], accHi[4];
        v8f zero = {};
#pragma unroll
        for (int q = 0; q < 4; ++q) { accLo[q] = zero; accHi[q] = zero; }

        const __bf16* arow = hS + (mt * 16 + ln) * H_STRIDE;
        for (int kt = 0; kt < 64; ++kt) {
            const __bf16* ap = arow + kt * 32 + 8 * g;
            v16bf a = cat8(*(const v8bf*)ap, *(const v8bf*)(ap + 16));
#pragma unroll
            for (int q = 0; q < 4; ++q) {
                int ntLo = half * 16 + pg * 4 + q;
                v16bf bLo = *(const v16bf*)(w2p + ((size_t)ntLo * 64 + kt) * 512 + lane * 16);
                v16bf bHi = *(const v16bf*)(w2p + ((size_t)(ntLo + 32) * 64 + kt) * 512 + lane * 16);
                accLo[q] = wmma_bf16(a, bLo, accLo[q]);
                accHi[q] = wmma_bf16(a, bHi, accHi[q]);
            }
        }
#pragma unroll
        for (int q = 0; q < 4; ++q) {
            int   ntLo = half * 16 + pg * 4 + q;
            int   o    = ntLo * 16 + ln;          // output column 0..511
            float blo  = b2[o];
            float bhi  = b2[OUTW + o];
            int   rbase = mt * 16 + 8 * g;
#pragma unroll
            for (int r = 0; r < 8; ++r) {
                int   row = rbase + r;
                float wv  = accLo[q][r] + blo;    // weight part
                float bv  = accHi[q][r] + bhi;    // bias part
                out[(size_t)(wgBase + row) * OUTW + o] = inpS[row] * wv + bv;
            }
        }
    }
}

extern "C" void kernel_launch(void* const* d_in, const int* in_sizes, int n_in,
                              void* d_out, int out_size, void* d_ws, size_t ws_size,
                              hipStream_t stream) {
    const float* x     = (const float*)d_in[0];
    const float* fc1_w = (const float*)d_in[1];
    const float* fc1_b = (const float*)d_in[2];
    const float* fc2_w = (const float*)d_in[3];
    const float* fc2_b = (const float*)d_in[4];
    const float* dil   = (const float*)d_in[5];
    const float* shp   = (const float*)d_in[6];
    float* out = (float*)d_out;

    // Workspace carve (≈4.5 MB)
    __bf16* w1p = (__bf16*)d_ws;                                   // 128*2048 bf16 = 512 KB
    __bf16* w2p = (__bf16*)((char*)d_ws + 524288);                 // 2048*1024 bf16 = 4 MB
    float*  b2f = (float*)((char*)d_ws + 524288 + 4194304);        // 1024 f32
    float*  vbuf = b2f + NFOLD;                                    // 2 f32

    compute_v_kernel<<<1, 64, 0, stream>>>(x, dil, shp, vbuf);
    pack_w1_kernel<<<(LAT * HID) / 256, 256, 0, stream>>>(fc1_w, w1p);
    pack_w2_kernel<<<(HID * NFOLD) / 256, 256, 0, stream>>>(fc2_w, vbuf, w2p);
    fold_b2_kernel<<<NFOLD / 256, 256, 0, stream>>>(fc2_b, vbuf, b2f);

    fused_gal_kernel<<<BATCH / 64, 256, SMEM_BYTES, stream>>>(x, w1p, fc1_b, w2p, b2f, out);
}
